// Decoder2_23261542875626
// MI455X (gfx1250) — matrix operand order repaired and checked by reading, not compiled, not run on hardware
//
#include <hip/hip_runtime.h>

#define H 512
#define BARS 16
#define UNITS 16
#define BATCH 256
#define K0 1536          // [out | emb | h0]
#define K1 1024          // [h0' | h1]
#define NG 2048          // 4*H gates
#define X0S 1544         // padded bf16 LDS row stride (bank-conflict free)
#define X1S 1032
#define CS  516          // padded f32 c-state row stride
#define KT0 48           // K0/32
#define KT1 32           // K1/32
#define KTO 16           // H/32
#define NT0 128          // NG/16
#define NTO 32           // H/16

typedef __attribute__((ext_vector_type(8)))  __bf16 v8bf;
typedef __attribute__((ext_vector_type(16))) __bf16 v16bf;
typedef __attribute__((ext_vector_type(8)))  float  v8f;

static __device__ __forceinline__ unsigned short f2bf(float f) {
  union { float f; unsigned u; } x; x.f = f;
  unsigned u = x.u;
  unsigned r = (u >> 16) & 1u;
  return (unsigned short)((u + 0x7fffu + r) >> 16);
}
static __device__ __forceinline__ float sigf(float x) {
  return 1.0f / (1.0f + __expf(-x));
}
static __device__ __forceinline__ float tanh_fast(float x) {
  return 2.0f / (1.0f + __expf(-2.0f * x)) - 1.0f;
}

// A fragment (16x32 bf16) from LDS row-major [16][stride] (padded stride).
static __device__ __forceinline__ v16bf load_a_lds(const unsigned short* base,
                                                   int stride, int kk, int lane) {
  const int m  = lane & 15;
  const int ko = (lane < 16) ? 0 : 16;
  const unsigned short* q = base + m * stride + kk + ko;
  v8bf lo = *reinterpret_cast<const v8bf*>(q);
  v8bf hi = *reinterpret_cast<const v8bf*>(q + 8);
  return __builtin_shufflevector(lo, hi, 0,1,2,3,4,5,6,7,8,9,10,11,12,13,14,15);
}

// B fragment from packed weights [kt][NT][32 lanes][16 bf16]: one coalesced
// 1024-byte wave read.
static __device__ __forceinline__ v16bf load_bp(const unsigned short* wp,
                                                int NT, int kt, int nt, int lane) {
  return *reinterpret_cast<const v16bf*>(wp + (((kt * NT + nt) * 32) + lane) * 16);
}

#define WMMA_BF16(A, B, C) \
  __builtin_amdgcn_wmma_f32_16x16x32_bf16(false, (A), false, (B), (short)0, (C), false, false)

// ---- weight prep: fp32 -> bf16, merge [W_ih|W_hh], fragment-pack for WMMA B ----
__global__ void prep_kernel(const float* __restrict__ Wih0, const float* __restrict__ Whh0,
                            const float* __restrict__ Wih1, const float* __restrict__ Whh1,
                            const float* __restrict__ Wout,
                            const float* __restrict__ bih0, const float* __restrict__ bhh0,
                            const float* __restrict__ bih1, const float* __restrict__ bhh1,
                            unsigned short* __restrict__ Wp0, unsigned short* __restrict__ Wp1,
                            unsigned short* __restrict__ Wpo,
                            float* __restrict__ bias0, float* __restrict__ bias1) {
  const int tid  = blockIdx.x * blockDim.x + threadIdx.x;
  const int nthr = gridDim.x * blockDim.x;
  for (int i = tid; i < KT0 * NT0 * 32 * 16; i += nthr) {
    int j = i & 15, lane = (i >> 4) & 31, nt = (i >> 9) & 127, kt = i >> 16;
    int n = nt * 16 + (lane & 15);
    int k = kt * 32 + ((lane & 16) ? 16 : 0) + j;
    float v = (k < 1024) ? Wih0[n * 1024 + k] : Whh0[n * 512 + (k - 1024)];
    Wp0[i] = f2bf(v);
  }
  for (int i = tid; i < KT1 * NT0 * 32 * 16; i += nthr) {
    int j = i & 15, lane = (i >> 4) & 31, nt = (i >> 9) & 127, kt = i >> 16;
    int n = nt * 16 + (lane & 15);
    int k = kt * 32 + ((lane & 16) ? 16 : 0) + j;
    float v = (k < 512) ? Wih1[n * 512 + k] : Whh1[n * 512 + (k - 512)];
    Wp1[i] = f2bf(v);
  }
  for (int i = tid; i < KTO * NTO * 32 * 16; i += nthr) {
    int j = i & 15, lane = (i >> 4) & 31, nt = (i >> 9) & 31, kt = i >> 14;
    int n = nt * 16 + (lane & 15);
    int k = kt * 32 + ((lane & 16) ? 16 : 0) + j;
    Wpo[i] = f2bf(Wout[n * 512 + k]);
  }
  for (int i = tid; i < NG; i += nthr) {
    bias0[i] = bih0[i] + bhh0[i];
    bias1[i] = bih1[i] + bhh1[i];
  }
}

__global__ __launch_bounds__(256, 1)
void decoder_kernel(const float* __restrict__ emb,     // [B, BARS, H]
                    const float* __restrict__ h0in,    // [BARS, 2, B, H]
                    const float* __restrict__ c0in,    // [BARS, 2, B, H]
                    const float* __restrict__ out0,    // [BARS, B, H]
                    const unsigned short* __restrict__ Wp0,
                    const unsigned short* __restrict__ Wp1,
                    const unsigned short* __restrict__ Wpo,
                    const float* __restrict__ bias0,
                    const float* __restrict__ bias1,
                    const float* __restrict__ bout,
                    float* __restrict__ dout)          // [B, BARS*UNITS, H]
{
  __shared__ unsigned short X0[32 * X0S];   // [out | emb | h0] bf16   (~97 KB)
  __shared__ unsigned short X1[32 * X1S];   // [h0' | h1] bf16         (~65 KB)
  __shared__ float C0s[32 * CS];            // layer0 cell state f32   (~65 KB)
  __shared__ float C1s[32 * CS];            // layer1 cell state f32   (~65 KB)
  __shared__ float redM[8 * 32];            // cross-wave max partials
  __shared__ float redS[8 * 32];            // cross-wave sum partials

  const int tid    = threadIdx.x;
  const int lane   = tid & 31;
  const int wv     = tid >> 5;            // wave 0..7 -> owns 64 hidden columns
  const int g      = blockIdx.x;          // 0..127 : 32-sequence tile
  const int bar    = g >> 3;
  const int btile  = (g & 7) * 32;
  const int lane15 = lane & 15;
  const int hi8    = (lane < 16) ? 0 : 8;

  // ---- initial state -> LDS ----
  for (int i = tid; i < 32 * H; i += 256) {
    int m = i >> 9, col = i & (H - 1);
    int b = btile + m;
    X0[m * X0S + col]        = f2bf(out0[bar * (BATCH * H) + b * H + col]);
    X0[m * X0S + 512 + col]  = f2bf(emb[b * (BARS * H) + bar * H + col]);
    X0[m * X0S + 1024 + col] = f2bf(h0in[bar * (2 * BATCH * H) + b * H + col]);
    X1[m * X1S + 512 + col]  = f2bf(h0in[bar * (2 * BATCH * H) + BATCH * H + b * H + col]);
    C0s[m * CS + col] = c0in[bar * (2 * BATCH * H) + b * H + col];
    C1s[m * CS + col] = c0in[bar * (2 * BATCH * H) + BATCH * H + b * H + col];
  }

  const v8f vzero = {0.f, 0.f, 0.f, 0.f, 0.f, 0.f, 0.f, 0.f};

  for (int u = 0; u < UNITS; ++u) {
    __syncthreads();   // (a) X0 consistent

    // ======== layer 0: one B fragment feeds 2 M-tiles; ping-pong k pipeline
    #pragma unroll
    for (int t = 0; t < 4; ++t) {
      v8f acc[2][4];
      #pragma unroll
      for (int mm = 0; mm < 2; ++mm)
        #pragma unroll
        for (int gt = 0; gt < 4; ++gt) acc[mm][gt] = vzero;
      const int ntb = wv * 4 + t;
      v16bf bA0 = load_bp(Wp0, NT0, 0, ntb,      lane);
      v16bf bA1 = load_bp(Wp0, NT0, 0, ntb + 32, lane);
      v16bf bA2 = load_bp(Wp0, NT0, 0, ntb + 64, lane);
      v16bf bA3 = load_bp(Wp0, NT0, 0, ntb + 96, lane);
      for (int kt = 0; kt < KT0; kt += 2) {
        const int k1 = kt + 1;
        v16bf bB0 = load_bp(Wp0, NT0, k1, ntb,      lane);
        v16bf bB1 = load_bp(Wp0, NT0, k1, ntb + 32, lane);
        v16bf bB2 = load_bp(Wp0, NT0, k1, ntb + 64, lane);
        v16bf bB3 = load_bp(Wp0, NT0, k1, ntb + 96, lane);
        v16bf a0 = load_a_lds(X0,            X0S, kt * 32, lane);
        v16bf a1 = load_a_lds(X0 + 16 * X0S, X0S, kt * 32, lane);
        acc[0][0] = WMMA_BF16(a0, bA0, acc[0][0]);
        acc[1][0] = WMMA_BF16(a1, bA0, acc[1][0]);
        acc[0][1] = WMMA_BF16(a0, bA1, acc[0][1]);
        acc[1][1] = WMMA_BF16(a1, bA1, acc[1][1]);
        acc[0][2] = WMMA_BF16(a0, bA2, acc[0][2]);
        acc[1][2] = WMMA_BF16(a1, bA2, acc[1][2]);
        acc[0][3] = WMMA_BF16(a0, bA3, acc[0][3]);
        acc[1][3] = WMMA_BF16(a1, bA3, acc[1][3]);
        const int k2 = (kt + 2 < KT0) ? kt + 2 : k1;
        bA0 = load_bp(Wp0, NT0, k2, ntb,      lane);
        bA1 = load_bp(Wp0, NT0, k2, ntb + 32, lane);
        bA2 = load_bp(Wp0, NT0, k2, ntb + 64, lane);
        bA3 = load_bp(Wp0, NT0, k2, ntb + 96, lane);
        a0 = load_a_lds(X0,            X0S, k1 * 32, lane);
        a1 = load_a_lds(X0 + 16 * X0S, X0S, k1 * 32, lane);
        acc[0][0] = WMMA_BF16(a0, bB0, acc[0][0]);
        acc[1][0] = WMMA_BF16(a1, bB0, acc[1][0]);
        acc[0][1] = WMMA_BF16(a0, bB1, acc[0][1]);
        acc[1][1] = WMMA_BF16(a1, bB1, acc[1][1]);
        acc[0][2] = WMMA_BF16(a0, bB2, acc[0][2]);
        acc[1][2] = WMMA_BF16(a1, bB2, acc[1][2]);
        acc[0][3] = WMMA_BF16(a0, bB3, acc[0][3]);
        acc[1][3] = WMMA_BF16(a1, bB3, acc[1][3]);
      }
      const int col = wv * 64 + t * 16 + lane15;
      const float bi = bias0[col],        bfv = bias0[512 + col];
      const float bg = bias0[1024 + col], bo  = bias0[1536 + col];
      #pragma unroll
      for (int mm = 0; mm < 2; ++mm) {
        #pragma unroll
        for (int e = 0; e < 8; ++e) {
          float iv = sigf(acc[mm][0][e] + bi);
          float fv = sigf(acc[mm][1][e] + bfv);
          float gv = tanh_fast(acc[mm][2][e] + bg);
          float ov = sigf(acc[mm][3][e] + bo);
          const int cidx = (mm * 16 + e + hi8) * CS + col;
          float c = fv * C0s[cidx] + iv * gv;
          C0s[cidx] = c;
          X1[(mm * 16 + e + hi8) * X1S + col] = f2bf(ov * tanh_fast(c)); // h0'
        }
      }
    }
    __syncthreads();   // (b) h0' complete; all X0 reads done

    // carry h0' into X0 h-slot for next step (8-byte LDS copies)
    for (int i = tid; i < 32 * 128; i += 256) {
      int m = i >> 7, col = (i & 127) * 4;
      *reinterpret_cast<uint2*>(&X0[m * X0S + 1024 + col]) =
          *reinterpret_cast<const uint2*>(&X1[m * X1S + col]);
    }

    // ======== layer 1
    unsigned int hp[2][4][4];   // packed bf16 pairs of h1' (deferred store)
    #pragma unroll
    for (int t = 0; t < 4; ++t) {
      v8f acc[2][4];
      #pragma unroll
      for (int mm = 0; mm < 2; ++mm)
        #pragma unroll
        for (int gt = 0; gt < 4; ++gt) acc[mm][gt] = vzero;
      const int ntb = wv * 4 + t;
      v16bf bA0 = load_bp(Wp1, NT0, 0, ntb,      lane);
      v16bf bA1 = load_bp(Wp1, NT0, 0, ntb + 32, lane);
      v16bf bA2 = load_bp(Wp1, NT0, 0, ntb + 64, lane);
      v16bf bA3 = load_bp(Wp1, NT0, 0, ntb + 96, lane);
      for (int kt = 0; kt < KT1; kt += 2) {
        const int k1 = kt + 1;
        v16bf bB0 = load_bp(Wp1, NT0, k1, ntb,      lane);
        v16bf bB1 = load_bp(Wp1, NT0, k1, ntb + 32, lane);
        v16bf bB2 = load_bp(Wp1, NT0, k1, ntb + 64, lane);
        v16bf bB3 = load_bp(Wp1, NT0, k1, ntb + 96, lane);
        v16bf a0 = load_a_lds(X1,            X1S, kt * 32, lane);
        v16bf a1 = load_a_lds(X1 + 16 * X1S, X1S, kt * 32, lane);
        acc[0][0] = WMMA_BF16(a0, bA0, acc[0][0]);
        acc[1][0] = WMMA_BF16(a1, bA0, acc[1][0]);
        acc[0][1] = WMMA_BF16(a0, bA1, acc[0][1]);
        acc[1][1] = WMMA_BF16(a1, bA1, acc[1][1]);
        acc[0][2] = WMMA_BF16(a0, bA2, acc[0][2]);
        acc[1][2] = WMMA_BF16(a1, bA2, acc[1][2]);
        acc[0][3] = WMMA_BF16(a0, bA3, acc[0][3]);
        acc[1][3] = WMMA_BF16(a1, bA3, acc[1][3]);
        const int k2 = (kt + 2 < KT1) ? kt + 2 : k1;
        bA0 = load_bp(Wp1, NT0, k2, ntb,      lane);
        bA1 = load_bp(Wp1, NT0, k2, ntb + 32, lane);
        bA2 = load_bp(Wp1, NT0, k2, ntb + 64, lane);
        bA3 = load_bp(Wp1, NT0, k2, ntb + 96, lane);
        a0 = load_a_lds(X1,            X1S, k1 * 32, lane);
        a1 = load_a_lds(X1 + 16 * X1S, X1S, k1 * 32, lane);
        acc[0][0] = WMMA_BF16(a0, bB0, acc[0][0]);
        acc[1][0] = WMMA_BF16(a1, bB0, acc[1][0]);
        acc[0][1] = WMMA_BF16(a0, bB1, acc[0][1]);
        acc[1][1] = WMMA_BF16(a1, bB1, acc[1][1]);
        acc[0][2] = WMMA_BF16(a0, bB2, acc[0][2]);
        acc[1][2] = WMMA_BF16(a1, bB2, acc[1][2]);
        acc[0][3] = WMMA_BF16(a0, bB3, acc[0][3]);
        acc[1][3] = WMMA_BF16(a1, bB3, acc[1][3]);
      }
      const int col = wv * 64 + t * 16 + lane15;
      const float bi = bias1[col],        bfv = bias1[512 + col];
      const float bg = bias1[1024 + col], bo  = bias1[1536 + col];
      #pragma unroll
      for (int mm = 0; mm < 2; ++mm) {
        unsigned short lo = 0;
        #pragma unroll
        for (int e = 0; e < 8; ++e) {
          float iv = sigf(acc[mm][0][e] + bi);
          float fv = sigf(acc[mm][1][e] + bfv);
          float gv = tanh_fast(acc[mm][2][e] + bg);
          float ov = sigf(acc[mm][3][e] + bo);
          const int cidx = (mm * 16 + e + hi8) * CS + col;
          float c = fv * C1s[cidx] + iv * gv;
          C1s[cidx] = c;
          unsigned short hb = f2bf(ov * tanh_fast(c));
          if ((e & 1) == 0) lo = hb;
          else hp[mm][t][e >> 1] = (unsigned)lo | ((unsigned)hb << 16);
        }
      }
    }
    __syncthreads();   // (c) all X1 reads (old h1) done

    #pragma unroll
    for (int t = 0; t < 4; ++t) {
      const int col = wv * 64 + t * 16 + lane15;
      #pragma unroll
      for (int mm = 0; mm < 2; ++mm) {
        #pragma unroll
        for (int p = 0; p < 4; ++p) {
          unsigned v = hp[mm][t][p];
          X1[(mm * 16 + 2 * p + hi8) * X1S + 512 + col]     = (unsigned short)v;
          X1[(mm * 16 + 2 * p + 1 + hi8) * X1S + 512 + col] = (unsigned short)(v >> 16);
        }
      }
    }
    __syncthreads();   // (d) h1' visible

    // ======== output projection: logits kept in registers
    v8f lg[4][2];
    #pragma unroll
    for (int t = 0; t < 4; ++t) {
      lg[t][0] = vzero; lg[t][1] = vzero;
      const int ntb = wv * 4 + t;
      v16bf bA0 = load_bp(Wpo, NTO, 0, ntb, lane);
      for (int kt = 0; kt < KTO; kt += 2) {
        const int k1 = kt + 1;
        v16bf bB0 = load_bp(Wpo, NTO, k1, ntb, lane);
        v16bf a0 = load_a_lds(X1,            X1S, 512 + kt * 32, lane);
        v16bf a1 = load_a_lds(X1 + 16 * X1S, X1S, 512 + kt * 32, lane);
        lg[t][0] = WMMA_BF16(a0, bA0, lg[t][0]);
        lg[t][1] = WMMA_BF16(a1, bA0, lg[t][1]);
        const int k2 = (kt + 2 < KTO) ? kt + 2 : k1;
        bA0 = load_bp(Wpo, NTO, k2, ntb, lane);
        a0 = load_a_lds(X1,            X1S, 512 + k1 * 32, lane);
        a1 = load_a_lds(X1 + 16 * X1S, X1S, 512 + k1 * 32, lane);
        lg[t][0] = WMMA_BF16(a0, bB0, lg[t][0]);
        lg[t][1] = WMMA_BF16(a1, bB0, lg[t][1]);
      }
      const float bb = bout[wv * 64 + t * 16 + lane15];
      #pragma unroll
      for (int mm = 0; mm < 2; ++mm)
        #pragma unroll
        for (int e = 0; e < 8; ++e)
          lg[t][mm][e] += bb;
    }

    // ======== register softmax: lane shfl reduce + 2-stage cross-wave reduce
    float mx[2][8];
    #pragma unroll
    for (int mm = 0; mm < 2; ++mm) {
      #pragma unroll
      for (int e = 0; e < 8; ++e) {
        float m = fmaxf(fmaxf(lg[0][mm][e], lg[1][mm][e]),
                        fmaxf(lg[2][mm][e], lg[3][mm][e]));
        #pragma unroll
        for (int off = 8; off >= 1; off >>= 1)
          m = fmaxf(m, __shfl_xor(m, off, 16));
        mx[mm][e] = m;
      }
    }
    if (lane15 == 0) {
      #pragma unroll
      for (int mm = 0; mm < 2; ++mm)
        #pragma unroll
        for (int e = 0; e < 8; ++e)
          redM[wv * 32 + mm * 16 + hi8 + e] = mx[mm][e];
    }
    __syncthreads();
    if (tid < 32) {
      float m = redM[tid];
      #pragma unroll
      for (int w = 1; w < 8; ++w) m = fmaxf(m, redM[w * 32 + tid]);
      redM[tid] = m;
    }
    __syncthreads();
    float sm[2][8];
    #pragma unroll
    for (int mm = 0; mm < 2; ++mm) {
      #pragma unroll
      for (int e = 0; e < 8; ++e) {
        const float rm = redM[mm * 16 + hi8 + e];
        float s = 0.f;
        #pragma unroll
        for (int t = 0; t < 4; ++t) {
          float ev = __expf(lg[t][mm][e] - rm);
          lg[t][mm][e] = ev;
          s += ev;
        }
        #pragma unroll
        for (int off = 8; off >= 1; off >>= 1)
          s += __shfl_xor(s, off, 16);
        sm[mm][e] = s;
      }
    }
    if (lane15 == 0) {
      #pragma unroll
      for (int mm = 0; mm < 2; ++mm)
        #pragma unroll
        for (int e = 0; e < 8; ++e)
          redS[wv * 32 + mm * 16 + hi8 + e] = sm[mm][e];
    }
    __syncthreads();
    if (tid < 32) {
      float s = redS[tid];
      #pragma unroll
      for (int w = 1; w < 8; ++w) s += redS[w * 32 + tid];
      redS[tid] = s;
    }
    __syncthreads();

    // write out: X0[:,0:512] bf16 (feedback) + dout fp32
    #pragma unroll
    for (int mm = 0; mm < 2; ++mm) {
      #pragma unroll
      for (int e = 0; e < 8; ++e) {
        const int row = mm * 16 + e + hi8;
        const float inv = 1.0f / redS[row];
        const int b = btile + row;
        float* orow = dout + (size_t)b * (BARS * UNITS * H) + (bar * UNITS + u) * H;
        #pragma unroll
        for (int t = 0; t < 4; ++t) {
          const int col = wv * 64 + t * 16 + lane15;
          float o = lg[t][mm][e] * inv;
          X0[row * X0S + col] = f2bf(o);
          orow[col] = o;
        }
      }
    }
  }
}

extern "C" void kernel_launch(void* const* d_in, const int* in_sizes, int n_in,
                              void* d_out, int out_size, void* d_ws, size_t ws_size,
                              hipStream_t stream) {
  const float* emb  = (const float*)d_in[0];
  // d_in[1] = input_x (unused by reference)
  const float* h0   = (const float*)d_in[2];
  const float* c0   = (const float*)d_in[3];
  const float* out0 = (const float*)d_in[4];
  const float* Wih0 = (const float*)d_in[5];
  const float* Whh0 = (const float*)d_in[6];
  const float* bih0 = (const float*)d_in[7];
  const float* bhh0 = (const float*)d_in[8];
  const float* Wih1 = (const float*)d_in[9];
  const float* Whh1 = (const float*)d_in[10];
  const float* bih1 = (const float*)d_in[11];
  const float* bhh1 = (const float*)d_in[12];
  const float* Wout = (const float*)d_in[13];
  const float* bout = (const float*)d_in[14];

  unsigned short* Wp0 = (unsigned short*)d_ws;                  // 48*128*32*16
  unsigned short* Wp1 = Wp0 + (size_t)KT0 * NT0 * 32 * 16;      // 32*128*32*16
  unsigned short* Wpo = Wp1 + (size_t)KT1 * NT0 * 32 * 16;      // 16*32*32*16
  float* bias0 = (float*)(Wpo + (size_t)KTO * NTO * 32 * 16);
  float* bias1 = bias0 + NG;

  prep_kernel<<<1024, 256, 0, stream>>>(Wih0, Whh0, Wih1, Whh1, Wout,
                                        bih0, bhh0, bih1, bhh1,
                                        Wp0, Wp1, Wpo, bias0, bias1);
  decoder_kernel<<<128, 256, 0, stream>>>(emb, h0, c0, out0,
                                          Wp0, Wp1, Wpo,
                                          bias0, bias1, bout,
                                          (float*)d_out);
}
